// MultiHeadCausalAttention_36893769073193
// MI455X (gfx1250) — matrix operand order repaired and checked by reading, not compiled, not run on hardware
//
#include <hip/hip_runtime.h>
#include <hip/hip_bf16.h>

// ---------------------------------------------------------------------------
// Multi-head causal attention forward, MI455X (gfx1250, wave32, WMMA).
// GEMMs: 128x128 LDS-tiled, K-step 64, double-buffered via
// global_load_async_to_lds_b128 (ASYNCcnt pipelining); 16 WMMA per barrier
// pair. Attention: flash-style streaming softmax, 64-wide KV blocks.
// ---------------------------------------------------------------------------

#define B_    2
#define T_    2048
#define DM_   1024
#define H_    16
#define HD_   64
#define ROWS_ (B_ * T_)          // 4096
#define SCALE_ 0.125f            // 1/sqrt(64)

typedef __attribute__((ext_vector_type(16))) _Float16 v16h;
typedef __attribute__((ext_vector_type(8)))  _Float16 v8h;
typedef __attribute__((ext_vector_type(8)))  float    v8f;

union V16U { v16h v; v8h h[2]; };

// ---- WMMA helpers ----------------------------------------------------------

__device__ __forceinline__ v8f wmma32(v16h a, v16h b, v8f c) {
  return __builtin_amdgcn_wmma_f32_16x16x32_f16(
      false, a, false, b, (short)0, c, false, false);
}

// A-matrix 16x32 fragment (contiguous k = 16*hi + s): lane = row (mod 16), hi = lane>>4.
__device__ __forceinline__ v16h load_a_frag(const _Float16* base, int lda,
                                            int k0, int lane) {
  int row = lane & 15;
  int hi  = lane >> 4;
  const _Float16* p = base + row * lda + k0 + hi * 16;
  V16U a;
  a.h[0] = *(const v8h*)(p);
  a.h[1] = *(const v8h*)(p + 8);
  return a.v;
}

// B-matrix 32x16 fragment: lane = column (mod 16), kg = lane>>4.
// `brow` = matrix whose row n IS column n of B (W in y=x*W^T, K in S=Q*K^T).
__device__ __forceinline__ v16h load_b_frag(const _Float16* brow, int ldb,
                                            int k0, int lane) {
  int col = lane & 15;
  int kg  = lane >> 4;
  return *(const v16h*)(brow + col * ldb + k0 + kg * 16);
}

// ---- CDNA5 async-to-LDS copy (ASYNCcnt data path) --------------------------

__device__ __forceinline__ void async_copy_b128(_Float16* lds_dst,
                                                const _Float16* gsrc) {
  unsigned loff = (unsigned)(size_t)lds_dst;
  asm volatile("global_load_async_to_lds_b128 %0, %1, off"
               :: "v"(loff), "v"(gsrc) : "memory");
}

__device__ __forceinline__ void wait_async_le8() {
  asm volatile("s_wait_asynccnt 0x8" ::: "memory");
}
__device__ __forceinline__ void wait_async_0() {
  asm volatile("s_wait_asynccnt 0x0" ::: "memory");
}

// ---- Kernel 1: fp32 -> fp16 convert ---------------------------------------

__global__ __launch_bounds__(256) void cvt_f32_f16_kernel(
    const float* __restrict__ src, _Float16* __restrict__ dst, int n) {
  int i = blockIdx.x * blockDim.x + threadIdx.x;
  if (i < n) dst[i] = (_Float16)src[i];
}

// ---- Tiled GEMM core: C(128x128) = A[m0:,K] * Bt[n0:,K]^T ------------------
// Block = 256 threads = 8 waves, wave (wm,wn) computes 32x64.
// As/Bs: double-buffered [2][128][64] f16 tiles staged with async-to-LDS.
// K-step 64: 16 WMMA between each barrier pair; last step peeled.

__device__ __forceinline__ void gemm_block_128(
    const _Float16* __restrict__ A, const _Float16* __restrict__ Bt,
    int lda, int ldb, int m0, int n0, int K,
    _Float16* As, _Float16* Bs, v8f acc[2][4]) {
  const int tid  = threadIdx.x;
  const int lane = tid & 31;
  const int wave = tid >> 5;
  const int wm = (wave & 3) * 32;   // wave row offset in 128-row block
  const int wn = (wave >> 2) * 64;  // wave col offset in 128-col block

#pragma unroll
  for (int mi = 0; mi < 2; ++mi)
#pragma unroll
    for (int nt = 0; nt < 4; ++nt)
      for (int r = 0; r < 8; ++r) acc[mi][nt][r] = 0.0f;

  const int Ksteps = K / 64;

  // stage tile (128 rows x 64 halves = 1024 b128 chunks; 4 per thread/matrix)
  auto stage = [&](int buf, int kb) {
    int k0 = kb * 64;
#pragma unroll
    for (int i = 0; i < 4; ++i) {
      int chunk = tid * 4 + i;          // 0..1023
      int row = chunk >> 3;             // 0..127
      int cc  = (chunk & 7) * 8;        // 0..56 halves
      async_copy_b128(As + buf * 8192 + row * 64 + cc,
                      A + (size_t)(m0 + row) * lda + k0 + cc);
      async_copy_b128(Bs + buf * 8192 + row * 64 + cc,
                      Bt + (size_t)(n0 + row) * ldb + k0 + cc);
    }
  };

  auto compute = [&](int buf) {
    const _Float16* as = As + buf * 8192;
    const _Float16* bs = Bs + buf * 8192;
#pragma unroll
    for (int kc = 0; kc < 2; ++kc) {
      v16h af0 = load_a_frag(as + (wm) * 64, 64, kc * 32, lane);
      v16h af1 = load_a_frag(as + (wm + 16) * 64, 64, kc * 32, lane);
#pragma unroll
      for (int nt = 0; nt < 4; ++nt) {
        v16h bf = load_b_frag(bs + (wn + nt * 16) * 64, 64, kc * 32, lane);
        acc[0][nt] = wmma32(af0, bf, acc[0][nt]);
        acc[1][nt] = wmma32(af1, bf, acc[1][nt]);
      }
    }
  };

  stage(0, 0);
  int buf = 0;
  for (int kb = 0; kb < Ksteps - 1; ++kb) {   // branch-free steady state
    stage(buf ^ 1, kb + 1);
    wait_async_le8();          // retire current stage (in-order completion)
    __syncthreads();
    compute(buf);
    __syncthreads();           // done reading buf before it is restaged
    buf ^= 1;
  }
  wait_async_0();              // peeled last step
  __syncthreads();
  compute(buf);
}

// ---- Kernel 2: QKV projections --------------------------------------------
// Grid (32, 8, 3), block 256. Writes f16 [B][H][T][64].

__global__ __launch_bounds__(256, 1) void proj_qkv_kernel(
    const _Float16* __restrict__ xh,
    const _Float16* __restrict__ wq,
    const _Float16* __restrict__ wk,
    const _Float16* __restrict__ wv,
    _Float16* __restrict__ qh,
    _Float16* __restrict__ kh,
    _Float16* __restrict__ vh) {
  __shared__ _Float16 As[2][128][64];
  __shared__ _Float16 Bs[2][128][64];

  const int lane = threadIdx.x & 31;
  const int wave = threadIdx.x >> 5;
  const int wm = (wave & 3) * 32;
  const int wn = (wave >> 2) * 64;
  const int m0 = blockIdx.x * 128;
  const int n0 = blockIdx.y * 128;
  const _Float16* W = (blockIdx.z == 0) ? wq : (blockIdx.z == 1) ? wk : wv;
  _Float16* dst     = (blockIdx.z == 0) ? qh : (blockIdx.z == 1) ? kh : vh;

  v8f acc[2][4];
  gemm_block_128(xh, W, DM_, DM_, m0, n0, DM_, &As[0][0][0], &Bs[0][0][0], acc);

  const int hi = lane >> 4;
  const int cl = lane & 15;
#pragma unroll
  for (int mi = 0; mi < 2; ++mi)
#pragma unroll
    for (int nt = 0; nt < 4; ++nt)
#pragma unroll
      for (int r = 0; r < 8; ++r) {
        int rowg = m0 + wm + mi * 16 + r + 8 * hi;
        int b = rowg >> 11, t = rowg & (T_ - 1);
        int j = n0 + wn + nt * 16 + cl;
        int h = j >> 6, d = j & 63;
        dst[(((b * H_ + h) * T_) + t) * HD_ + d] = (_Float16)acc[mi][nt][r];
      }
}

// ---- Kernel 3: flash attention --------------------------------------------
// Grid (T/64, B*H), block 128 (4 waves, one 16-row Q tile each).
// 64-wide KV blocks: 16 WMMA per block, 2 barriers per block.

__global__ __launch_bounds__(128, 1) void flash_attn_kernel(
    const _Float16* __restrict__ qg,
    const _Float16* __restrict__ kg_,
    const _Float16* __restrict__ vg,
    _Float16* __restrict__ attn) {
  __shared__ _Float16 Vt[HD_][64];       // V block transposed: Vt[d][kk], 8KB
  __shared__ _Float16 Pt[4][16][64];     // per-wave P tile (C->A relayout), 8KB

  const int lane = threadIdx.x & 31;
  const int wave = threadIdx.x >> 5;
  const int bh = blockIdx.y;
  const int b = bh >> 4, h = bh & 15;
  const int wgT0 = blockIdx.x * 64;
  const int t0 = wgT0 + wave * 16;

  const _Float16* qh = qg + ((b * H_ + h) * T_) * HD_;
  const _Float16* kh = kg_ + ((b * H_ + h) * T_) * HD_;
  const _Float16* vh = vg + ((b * H_ + h) * T_) * HD_;

  v16h qa[2];
  qa[0] = load_a_frag(qh + t0 * HD_, HD_, 0, lane);
  qa[1] = load_a_frag(qh + t0 * HD_, HD_, 32, lane);

  float mrow[8], lrow[8];
  v8f o[4];
#pragma unroll
  for (int r = 0; r < 8; ++r) { mrow[r] = -1e30f; lrow[r] = 0.0f; }
#pragma unroll
  for (int nt = 0; nt < 4; ++nt)
    for (int r = 0; r < 8; ++r) o[nt][r] = 0.0f;

  const int hi = lane >> 4;
  const int cl = lane & 15;
  const int kvEnd = wgT0 + 64;

  for (int j0 = 0; j0 < kvEnd; j0 += 64) {
    __syncthreads();                      // Vt safe to overwrite
    {   // stage 64x64 V block transposed (each thread: 32 halves)
      int kk = threadIdx.x >> 1;          // 0..63
      int d0 = (threadIdx.x & 1) * 32;    // 0 or 32
      v16h vv0 = *(const v16h*)(vh + (j0 + kk) * HD_ + d0);
      v16h vv1 = *(const v16h*)(vh + (j0 + kk) * HD_ + d0 + 16);
#pragma unroll
      for (int i = 0; i < 16; ++i) {
        Vt[d0 + i][kk]      = vv0[i];
        Vt[d0 + 16 + i][kk] = vv1[i];
      }
    }
    __syncthreads();

    int act = __builtin_amdgcn_readfirstlane((j0 <= t0 + 15) ? 1 : 0);

    if (act) {
      // S(16x64) = Q(16x64) @ K_block^T : four 16x16 C frags
      v8f s[4];
#pragma unroll
      for (int nt = 0; nt < 4; ++nt)
        for (int r = 0; r < 8; ++r) s[nt][r] = 0.0f;
#pragma unroll
      for (int kc = 0; kc < 2; ++kc)
#pragma unroll
        for (int nt = 0; nt < 4; ++nt) {
          v16h bk = load_b_frag(kh + (j0 + nt * 16) * HD_, HD_, kc * 32, lane);
          s[nt] = wmma32(qa[kc], bk, s[nt]);
        }

      // online softmax (row = 16 lanes of one VGPR in C layout)
#pragma unroll
      for (int r = 0; r < 8; ++r) {
        int rowg = t0 + r + 8 * hi;
        float sv[4];
        float rv = -1e30f;
#pragma unroll
        for (int nt = 0; nt < 4; ++nt) {
          int c = j0 + nt * 16 + cl;
          float v = s[nt][r] * SCALE_;
          if (c > rowg) v = -1e30f;
          sv[nt] = v;
          rv = fmaxf(rv, v);
        }
        for (int off = 1; off < 16; off <<= 1)
          rv = fmaxf(rv, __shfl_xor(rv, off, 32));
        float mnew = fmaxf(mrow[r], rv);
        float alpha = __expf(mrow[r] - mnew);
        float ps = 0.0f;
#pragma unroll
        for (int nt = 0; nt < 4; ++nt) {
          int c = j0 + nt * 16 + cl;
          float p = (c > rowg) ? 0.0f : __expf(sv[nt] - mnew);
          ps += p;
          Pt[wave][r + 8 * hi][nt * 16 + cl] = (_Float16)p;
        }
        for (int off = 1; off < 16; off <<= 1)
          ps += __shfl_xor(ps, off, 32);
        lrow[r] = lrow[r] * alpha + ps;
        mrow[r] = mnew;
#pragma unroll
        for (int nt = 0; nt < 4; ++nt) o[nt][r] *= alpha;
      }

      // O(16x64) += P(16x64) @ V_block(64x64); Pt is wave-private, LDS ops
      // are in-order within a wave -> no barrier needed for the round-trip.
      v16h pa0 = load_a_frag(&Pt[wave][0][0], 64, 0, lane);
      v16h pa1 = load_a_frag(&Pt[wave][0][0], 64, 32, lane);
#pragma unroll
      for (int nt = 0; nt < 4; ++nt) {
        v16h bv0 = *(const v16h*)(&Vt[nt * 16 + cl][hi * 16]);
        v16h bv1 = *(const v16h*)(&Vt[nt * 16 + cl][32 + hi * 16]);
        o[nt] = wmma32(pa0, bv0, o[nt]);
        o[nt] = wmma32(pa1, bv1, o[nt]);
      }
    }
  }

  // normalize and write attn output as f16 in [B][T][H*64] (= GEMM-ready)
#pragma unroll
  for (int nt = 0; nt < 4; ++nt)
#pragma unroll
    for (int r = 0; r < 8; ++r) {
      int t = t0 + r + 8 * hi;
      int col = h * HD_ + nt * 16 + cl;
      float val = o[nt][r] / lrow[r];
      attn[(b * T_ + t) * DM_ + col] = (_Float16)val;
    }
}

// ---- Kernel 4: output projection  out = attn @ Wo^T (f32 out) -------------

__global__ __launch_bounds__(256, 1) void out_proj_kernel(
    const _Float16* __restrict__ ah,
    const _Float16* __restrict__ wo,
    float* __restrict__ out) {
  __shared__ _Float16 As[2][128][64];
  __shared__ _Float16 Bs[2][128][64];

  const int lane = threadIdx.x & 31;
  const int wave = threadIdx.x >> 5;
  const int wm = (wave & 3) * 32;
  const int wn = (wave >> 2) * 64;
  const int m0 = blockIdx.x * 128;
  const int n0 = blockIdx.y * 128;

  v8f acc[2][4];
  gemm_block_128(ah, wo, DM_, DM_, m0, n0, DM_, &As[0][0][0], &Bs[0][0][0], acc);

  const int hi = lane >> 4;
  const int cl = lane & 15;
#pragma unroll
  for (int mi = 0; mi < 2; ++mi)
#pragma unroll
    for (int nt = 0; nt < 4; ++nt)
#pragma unroll
      for (int r = 0; r < 8; ++r) {
        int rowg = m0 + wm + mi * 16 + r + 8 * hi;
        int j = n0 + wn + nt * 16 + cl;
        out[(size_t)rowg * DM_ + j] = acc[mi][nt][r];
      }
}

// ---- Host-side launch ------------------------------------------------------

extern "C" void kernel_launch(void* const* d_in, const int* in_sizes, int n_in,
                              void* d_out, int out_size, void* d_ws, size_t ws_size,
                              hipStream_t stream) {
  const float* x  = (const float*)d_in[0];
  const float* Wq = (const float*)d_in[1];
  const float* Wk = (const float*)d_in[2];
  const float* Wv = (const float*)d_in[3];
  const float* Wo = (const float*)d_in[4];
  float* out = (float*)d_out;

  const int XN = ROWS_ * DM_;    // 4,194,304
  const int WN = DM_ * DM_;      // 1,048,576

  _Float16* w = (_Float16*)d_ws;
  _Float16* xh  = w;                // [4096][1024]
  _Float16* wqh = w + (size_t)XN;
  _Float16* wkh = wqh + WN;
  _Float16* wvh = wkh + WN;
  _Float16* woh = wvh + WN;
  _Float16* qh  = woh + WN;         // [B][H][T][64]
  _Float16* kh  = qh + (size_t)XN;
  _Float16* vh  = kh + (size_t)XN;
  _Float16* ath = vh + (size_t)XN;  // [B][T][H*64]

  cvt_f32_f16_kernel<<<(XN + 255) / 256, 256, 0, stream>>>(x, xh, XN);
  cvt_f32_f16_kernel<<<(WN + 255) / 256, 256, 0, stream>>>(Wq, wqh, WN);
  cvt_f32_f16_kernel<<<(WN + 255) / 256, 256, 0, stream>>>(Wk, wkh, WN);
  cvt_f32_f16_kernel<<<(WN + 255) / 256, 256, 0, stream>>>(Wv, wvh, WN);
  cvt_f32_f16_kernel<<<(WN + 255) / 256, 256, 0, stream>>>(Wo, woh, WN);

  proj_qkv_kernel<<<dim3(ROWS_ / 128, DM_ / 128, 3), 256, 0, stream>>>(
      xh, wqh, wkh, wvh, qh, kh, vh);

  flash_attn_kernel<<<dim3(T_ / 64, B_ * H_), 128, 0, stream>>>(
      qh, kh, vh, ath);

  out_proj_kernel<<<dim3(ROWS_ / 128, DM_ / 128), 256, 0, stream>>>(
      ath, woh, out);
}
